// DepthAugmentedBEVLifter_14156212208360
// MI455X (gfx1250) — matrix operand order repaired and checked by reading, not compiled, not run on hardware
//
#include <hip/hip_runtime.h>
#include <hip/hip_bf16.h>

// ---- WMMA fragment types (wave32, CDNA5) ----
typedef __attribute__((ext_vector_type(16))) _Float16 v16h;
typedef __attribute__((ext_vector_type(8)))  float    v8f;

// Problem constants (from reference)
#define BATCH   12
#define H1W1    45056            // 128*352
#define N5      11264            // 64*176
#define W5      176
#define OUT_MAIN ((size_t)BATCH * 128 * 128 * 128)   // (12,128,128,128)
#define BEV_ELEMS (BATCH * 128 * 128 * 64)

// A-matrix (16x32 f16) K index pattern for vector element e in lane-half `hl`
// (ISA 7.12.2: lanes 0-15 hold K{0-7,16-23}, lanes 16-31 hold K{8-15,24-31})
__device__ __forceinline__ int a_kpat(int e, int hl) {
  return ((e & 8) ? 16 : 0) + hl * 8 + (e & 7);
}

// ---------------- Kernel 0: zero BEV accumulator in workspace ----------------
__global__ void k_zero(float* __restrict__ p, int n) {
  int i = blockIdx.x * blockDim.x + threadIdx.x;
  if (i < n) p[i] = 0.0f;
}

// ---------------- Kernel 1: stage1 -> r1 -> skip (fused dual 1x1 conv) -------
__global__ __launch_bounds__(256) void k_skip(
    const float* __restrict__ stage1,
    const float* __restrict__ red1_w, const float* __restrict__ red1_g, const float* __restrict__ red1_b,
    const float* __restrict__ skip_w, const float* __restrict__ skip_g, const float* __restrict__ skip_b,
    float* __restrict__ skip_out) {
  __shared__ _Float16 w1[64 * 64];        // red1_w as f16
  __shared__ _Float16 w2[32 * 64];        // skip_w as f16
  __shared__ _Float16 r1s[8 * 64 * 16];   // per-wave r1 tile (64ch x 16px) f16

  const int tid = threadIdx.x;
  for (int i = tid; i < 64 * 64; i += 256) w1[i] = (_Float16)red1_w[i];
  for (int i = tid; i < 32 * 64; i += 256) w2[i] = (_Float16)skip_w[i];
  __syncthreads();

  const int b    = blockIdx.x / 352;
  const int pb   = blockIdx.x % 352;
  const int wave = tid >> 5;
  const int lane = tid & 31;
  const int p0   = pb * 128 + wave * 16;
  const int ml   = lane & 15;
  const int hl   = lane >> 4;

  // B fragments from stage1: lane = pixel, elements = K (channels) in a_kpat order
  const float* s1b = stage1 + (size_t)b * 64 * H1W1;
  v16h bf[2];
  #pragma unroll
  for (int k = 0; k < 2; ++k) {
    const float* src = s1b + p0 + ml;
    #pragma unroll
    for (int e = 0; e < 16; ++e) bf[k][e] = (_Float16)src[(size_t)(k * 32 + a_kpat(e, hl)) * H1W1];
  }

  // GEMM 1: r1(64x16) = red1_w(64x64) * stage1(64x16), BN+ReLU, park in LDS
  #pragma unroll
  for (int g = 0; g < 4; ++g) {
    v8f acc = {};
    #pragma unroll
    for (int k = 0; k < 2; ++k) {
      v16h af;
      const int o = g * 16 + ml;
      #pragma unroll
      for (int e = 0; e < 16; ++e) af[e] = w1[o * 64 + k * 32 + a_kpat(e, hl)];
      acc = __builtin_amdgcn_wmma_f32_16x16x32_f16(false, af, false, bf[k],
                                                   (short)0, acc, false, false);
    }
    #pragma unroll
    for (int r = 0; r < 8; ++r) {
      const int o = g * 16 + r + 8 * hl;
      float v = acc[r] * red1_g[o] + red1_b[o];
      v = v > 0.0f ? v : 0.0f;
      r1s[(wave * 64 + o) * 16 + ml] = (_Float16)v;   // D-layout -> LDS
    }
  }
  __syncthreads();

  // GEMM 2: skip(32x16) = skip_w(32x64) * r1(64x16)
  v16h b2[2];
  #pragma unroll
  for (int k = 0; k < 2; ++k) {
    const _Float16* src = &r1s[wave * 64 * 16 + ml];
    #pragma unroll
    for (int e = 0; e < 16; ++e) b2[k][e] = src[(k * 32 + a_kpat(e, hl)) * 16];
  }
  float* skb = skip_out + (size_t)b * 32 * H1W1;
  #pragma unroll
  for (int g = 0; g < 2; ++g) {
    v8f acc = {};
    #pragma unroll
    for (int k = 0; k < 2; ++k) {
      v16h af;
      const int o = g * 16 + ml;
      #pragma unroll
      for (int e = 0; e < 16; ++e) af[e] = w2[o * 64 + k * 32 + a_kpat(e, hl)];
      acc = __builtin_amdgcn_wmma_f32_16x16x32_f16(false, af, false, b2[k],
                                                   (short)0, acc, false, false);
    }
    #pragma unroll
    for (int r = 0; r < 8; ++r) {
      const int o = g * 16 + r + 8 * hl;
      float v = acc[r] * skip_g[o] + skip_b[o];
      v = v > 0.0f ? v : 0.0f;
      skb[(size_t)o * H1W1 + p0 + ml] = v;
    }
  }
}

// -------- Kernel 2: stage5 -> {reduced, depth} -> geometry -> BEV scatter ----
__global__ __launch_bounds__(256) void k_lift(
    const float* __restrict__ stage5,
    const float* __restrict__ intr, const float* __restrict__ extr,
    const float* __restrict__ red5_w, const float* __restrict__ red5_g, const float* __restrict__ red5_b,
    const float* __restrict__ dep5_w, const float* __restrict__ dep5_g, const float* __restrict__ dep5_b,
    float* __restrict__ bev) {
  __shared__ _Float16 wl[(64 + 48) * 256];   // half-K slab of red5_w + dep5_w (56 KB)

  const int tid  = threadIdx.x;
  const int wave = tid >> 5;
  const int lane = tid & 31;
  const int b  = blockIdx.x / 88;
  const int pb = blockIdx.x % 88;
  const int p0 = pb * 128 + wave * 16;
  const int ml = lane & 15;
  const int hl = lane >> 4;

  const float* s5b = stage5 + (size_t)b * 512 * N5;
  v8f accR[4] = {};   // reduced (64 ch)
  v8f accL[3] = {};   // depth logits (48 ch)

  for (int kh = 0; kh < 2; ++kh) {           // K=512 in two 256-wide slabs
    __syncthreads();
    for (int i = tid; i < 64 * 256; i += 256) {
      const int o = i >> 8, k = i & 255;
      wl[i] = (_Float16)red5_w[o * 512 + kh * 256 + k];
    }
    for (int i = tid; i < 48 * 256; i += 256) {
      const int o = i >> 8, k = i & 255;
      wl[64 * 256 + i] = (_Float16)dep5_w[o * 512 + kh * 256 + k];
    }
    __syncthreads();

    for (int ks = 0; ks < 8; ++ks) {
      const int c = kh * 256 + ks * 32;
      const float* src = s5b + (size_t)c * N5 + p0 + ml;
      if (ks + 1 < 8) __builtin_prefetch(src + 32 * N5, 0, 1);  // global_prefetch_b8
      v16h bf;
      #pragma unroll
      for (int e = 0; e < 16; ++e) bf[e] = (_Float16)src[(size_t)a_kpat(e, hl) * N5];
      #pragma unroll
      for (int g = 0; g < 4; ++g) {
        v16h af;
        const int o = g * 16 + ml;
        #pragma unroll
        for (int e = 0; e < 16; ++e) af[e] = wl[o * 256 + ks * 32 + a_kpat(e, hl)];
        accR[g] = __builtin_amdgcn_wmma_f32_16x16x32_f16(false, af, false, bf,
                                                         (short)0, accR[g], false, false);
      }
      #pragma unroll
      for (int g = 0; g < 3; ++g) {
        v16h af;
        const int o = g * 16 + ml;
        #pragma unroll
        for (int e = 0; e < 16; ++e)
          af[e] = wl[64 * 256 + o * 256 + ks * 32 + a_kpat(e, hl)];
        accL[g] = __builtin_amdgcn_wmma_f32_16x16x32_f16(false, af, false, bf,
                                                         (short)0, accL[g], false, false);
      }
    }
  }

  // Softmax over 48 bins (channels split across lane-halves n and n+16)
  float lg[24];
  float mx = -1e30f;
  #pragma unroll
  for (int g = 0; g < 3; ++g)
    #pragma unroll
    for (int r = 0; r < 8; ++r) {
      const int o = g * 16 + r + 8 * hl;
      const float l = (accL[g][r] * dep5_g[o] + dep5_b[o]) * 10.0f;
      lg[g * 8 + r] = l;
      mx = fmaxf(mx, l);
    }
  mx = fmaxf(mx, __shfl_xor(mx, 16, 32));
  const float kb = 0.08711371f;  // ln(60)/47, bins = exp(o*kb)
  float se = 0.0f, sb = 0.0f;
  #pragma unroll
  for (int g = 0; g < 3; ++g)
    #pragma unroll
    for (int r = 0; r < 8; ++r) {
      const int o = g * 16 + r + 8 * hl;
      const float e = __expf(lg[g * 8 + r] - mx);
      se += e;
      sb += e * __expf((float)o * kb);
    }
  se += __shfl_xor(se, 16, 32);
  sb += __shfl_xor(sb, 16, 32);
  float depth = fminf(fmaxf(sb / se, 1.0f), 65.0f);

  // Pixel -> ego geometry
  const int p = p0 + ml;
  const float px = (float)(p % W5), py = (float)(p / W5);
  const float* Ki = intr + b * 9;
  const float a00 = Ki[0], a01 = Ki[1], a02 = Ki[2];
  const float a10 = Ki[3], a11 = Ki[4], a12 = Ki[5];
  const float a20 = Ki[6], a21 = Ki[7], a22 = Ki[8];
  const float det = a00 * (a11 * a22 - a12 * a21) - a01 * (a10 * a22 - a12 * a20)
                  + a02 * (a10 * a21 - a11 * a20);
  const float id = 1.0f / det;
  const float i00 = (a11 * a22 - a12 * a21) * id, i01 = (a02 * a21 - a01 * a22) * id,
              i02 = (a01 * a12 - a02 * a11) * id;
  const float i10 = (a12 * a20 - a10 * a22) * id, i11 = (a00 * a22 - a02 * a20) * id,
              i12 = (a02 * a10 - a00 * a12) * id;
  const float i20 = (a10 * a21 - a11 * a20) * id, i21 = (a01 * a20 - a00 * a21) * id,
              i22 = (a00 * a11 - a01 * a10) * id;
  const float cx = depth * (i00 * px + i01 * py + i02);
  const float cy = depth * (i10 * px + i11 * py + i12);
  const float cz = depth * (i20 * px + i21 * py + i22);
  const float* E = extr + b * 16;
  const float ex = E[0] * cx + E[1] * cy + E[2]  * cz + E[3];
  const float ey = E[4] * cx + E[5] * cy + E[6]  * cz + E[7];
  const float ez = E[8] * cx + E[9] * cy + E[10] * cz + E[11];
  const bool ev = (ex >= -51.2f) && (ex < 51.2f) && (ey >= -51.2f) && (ey < 51.2f)
               && (ez >= -5.0f) && (ez < 3.0f);
  const int bx = (int)floorf(ex * 2.5f + 64.0f);
  const int by = (int)floorf(ey * 2.5f + 64.0f);
  const bool valid = ev && (bx >= 0) && (bx < 128) && (by >= 0) && (by < 128);
  const float wz = __expf(-0.05f * fabsf(ez));

  if (valid) {
    float* dst = bev + ((size_t)b * 16384 + (size_t)(by * 128 + bx)) * 64;
    #pragma unroll
    for (int g = 0; g < 4; ++g)
      #pragma unroll
      for (int r = 0; r < 8; ++r) {
        const int o = g * 16 + r + 8 * hl;
        float v = accR[g][r] * red5_g[o] + red5_b[o];
        v = v > 0.0f ? v : 0.0f;
        atomicAdd(dst + o, v * wz);   // global_atomic_add_f32
      }
  }
}

// ---------------- Kernel 3: 3x3 conv on BEV (implicit GEMM) ------------------
__global__ __launch_bounds__(256) void k_conv(
    const float* __restrict__ bev,
    const float* __restrict__ main_w, const float* __restrict__ main_g,
    const float* __restrict__ main_b, float* __restrict__ out) {
  __shared__ _Float16 wl[64 * 3 * 64];   // 64 out x 3 taps x 64 in, f16 (24 KB)

  const int tid  = threadIdx.x;
  const int wave = tid >> 5;
  const int lane = tid & 31;
  const int bid  = blockIdx.x;            // ((b*128)+y)*2 + half_o
  const int half_o = bid & 1;
  const int y  = (bid >> 1) & 127;
  const int b  = bid >> 8;
  const int x0 = wave * 16;
  const int ml = lane & 15;
  const int hl = lane >> 4;

  const float* bevb = bev + (size_t)b * 16384 * 64;
  v8f acc[4] = {};

  for (int tg = 0; tg < 3; ++tg) {        // taps in groups of 3
    __syncthreads();
    for (int i = tid; i < 64 * 3 * 64; i += 256) {
      const int ol = i / 192, t = (i / 64) % 3, c = i % 64;
      const int tap = tg * 3 + t;
      const int o = half_o * 64 + ol;
      wl[i] = (_Float16)main_w[(o * 64 + c) * 9 + tap];
    }
    __syncthreads();

    for (int t = 0; t < 3; ++t) {
      const int tap = tg * 3 + t;
      const int ky = tap / 3, kx = tap % 3;
      const int yy = y + ky - 1;
      const bool yok = (yy >= 0) && (yy < 128);
      #pragma unroll
      for (int ks = 0; ks < 2; ++ks) {
        v16h bf;
        const int xx = x0 + ml + kx - 1;
        #pragma unroll
        for (int e = 0; e < 16; ++e) {
          const int c = ks * 32 + a_kpat(e, hl);
          float v = 0.0f;
          if (yok && xx >= 0 && xx < 128)
            v = bevb[((size_t)yy * 128 + xx) * 64 + c];
          bf[e] = (_Float16)v;
        }
        #pragma unroll
        for (int g = 0; g < 4; ++g) {
          v16h af;
          const int ol = g * 16 + ml;
          #pragma unroll
          for (int e = 0; e < 16; ++e)
            af[e] = wl[ol * 192 + t * 64 + ks * 32 + a_kpat(e, hl)];
          acc[g] = __builtin_amdgcn_wmma_f32_16x16x32_f16(false, af, false, bf,
                                                          (short)0, acc[g], false, false);
        }
      }
    }
  }

  float* ob = out + (size_t)b * 128 * 16384;
  #pragma unroll
  for (int g = 0; g < 4; ++g)
    #pragma unroll
    for (int r = 0; r < 8; ++r) {
      const int o = half_o * 64 + g * 16 + r + 8 * hl;
      float v = acc[g][r] * main_g[o] + main_b[o];
      v = v > 0.0f ? v : 0.0f;
      ob[(size_t)o * 16384 + y * 128 + x0 + ml] = v;
    }
}

// ---------------------------------------------------------------------------
extern "C" void kernel_launch(void* const* d_in, const int* in_sizes, int n_in,
                              void* d_out, int out_size, void* d_ws, size_t ws_size,
                              hipStream_t stream) {
  const float* stage1 = (const float*)d_in[0];
  const float* stage5 = (const float*)d_in[1];
  const float* intr   = (const float*)d_in[2];
  const float* extr   = (const float*)d_in[3];
  const float* red1_w = (const float*)d_in[4];
  const float* red1_g = (const float*)d_in[5];
  const float* red1_b = (const float*)d_in[6];
  const float* skip_w = (const float*)d_in[7];
  const float* skip_g = (const float*)d_in[8];
  const float* skip_b = (const float*)d_in[9];
  const float* red5_w = (const float*)d_in[10];
  const float* red5_g = (const float*)d_in[11];
  const float* red5_b = (const float*)d_in[12];
  const float* dep5_w = (const float*)d_in[13];
  const float* dep5_g = (const float*)d_in[14];
  const float* dep5_b = (const float*)d_in[15];
  const float* main_w = (const float*)d_in[16];
  const float* main_g = (const float*)d_in[17];
  const float* main_b = (const float*)d_in[18];

  float* outp = (float*)d_out;                 // (12,128,128,128) NCHW
  float* skip = outp + OUT_MAIN;               // (12,32,128,352)  NCHW
  float* bev  = (float*)d_ws;                  // (12,128,128,64)  NHWC accumulator

  k_zero<<<(BEV_ELEMS + 255) / 256, 256, 0, stream>>>(bev, BEV_ELEMS);
  k_skip<<<BATCH * 352, 256, 0, stream>>>(stage1, red1_w, red1_g, red1_b,
                                          skip_w, skip_g, skip_b, skip);
  k_lift<<<BATCH * 88, 256, 0, stream>>>(stage5, intr, extr,
                                         red5_w, red5_g, red5_b,
                                         dep5_w, dep5_g, dep5_b, bev);
  k_conv<<<BATCH * 128 * 2, 256, 0, stream>>>(bev, main_w, main_g, main_b, outp);
}
